// TopKGate_34102040330679
// MI455X (gfx1250) — matrix operand order repaired and checked by reading, not compiled, not run on hardware
//
#include <hip/hip_runtime.h>
#include <hip/hip_fp16.h>

typedef __attribute__((ext_vector_type(16))) _Float16 v16h;
typedef __attribute__((ext_vector_type(8)))  float    v8f;

#define N_TOK   262144
#define DIM     1024
#define NEXP    64
#define TOPK    8
#define ROWS_PB 128      // rows per block (8 waves x 16 rows)
#define LDS_STR 65       // padded row stride in LDS (bank-conflict-free)

// ---------------- prep kernels ----------------

__global__ __launch_bounds__(64) void zero_counts_k(unsigned* counts) {
    counts[threadIdx.x] = 0u;
}

__global__ __launch_bounds__(256) void convert_w_k(const float* __restrict__ W,
                                                   _Float16* __restrict__ Wh) {
    int i = blockIdx.x * 256 + threadIdx.x;   // grid covers NEXP*DIM = 65536
    Wh[i] = (_Float16)W[i];
}

// ---------------- main fused gate kernel ----------------

__global__ __launch_bounds__(256) void moe_gate_k(const float* __restrict__ x,
                                                  const _Float16* __restrict__ Wh,
                                                  const float* __restrict__ bias,
                                                  float* __restrict__ out_idx,
                                                  float* __restrict__ out_gate,
                                                  unsigned* __restrict__ g_counts) {
    __shared__ float    logits_s[ROWS_PB * LDS_STR];
    __shared__ unsigned hist_s[NEXP];

    const int tid  = threadIdx.x;
    const int wave = tid >> 5;          // 0..7
    const int lane = tid & 31;
    const int lo   = lane & 15;         // 0..15
    const int hi   = lane >> 4;         // 0..1

    if (tid < NEXP) hist_s[tid] = 0u;

    // This lane's A-matrix row (A layout: M = lane%16, K-halves by lane/16)
    const int row  = blockIdx.x * ROWS_PB + wave * 16 + lo;
    const float* xrow = x + (size_t)row * DIM;

    // bias for the 4 N-tiles this lane touches in C (N = tile*16 + lo)
    const float bb0 = bias[lo];
    const float bb1 = bias[16 + lo];
    const float bb2 = bias[32 + lo];
    const float bb3 = bias[48 + lo];

    v8f c0 = {}, c1 = {}, c2 = {}, c3 = {};

    for (int k0 = 0; k0 < DIM; k0 += 32) {
        // ---- A fragment: 16 f16 per lane, converted in-register from f32 x ----
        union { float4 f[4]; float s[16]; } af;
        const float4* xa  = (const float4*)(xrow + k0 + hi * 16);
        const float4* xa2 = (const float4*)(xrow + k0 + hi * 16 + 8);
        af.f[0] = xa[0];  af.f[1] = xa[1];    // K = k0 + hi*16 + 0..7
        af.f[2] = xa2[0]; af.f[3] = xa2[1];   // K = k0 + hi*16 + 8..15
        if (k0 + 32 < DIM) __builtin_prefetch(xrow + k0 + 32 + hi * 8, 0, 1);

        union { v16h v; _Float16 h[16]; } Au;
#pragma unroll
        for (int i = 0; i < 16; ++i) Au.h[i] = (_Float16)af.s[i];

        // ---- B fragments: lane n = tile*16 + lo, 16 contiguous f16 (K = k0 + hi*16 .. +15)
        const _Float16* wb = Wh + k0 + hi * 16;
        v16h B0 = *(const v16h*)(wb + (size_t)(lo)      * DIM);
        v16h B1 = *(const v16h*)(wb + (size_t)(16 + lo) * DIM);
        v16h B2 = *(const v16h*)(wb + (size_t)(32 + lo) * DIM);
        v16h B3 = *(const v16h*)(wb + (size_t)(48 + lo) * DIM);

        c0 = __builtin_amdgcn_wmma_f32_16x16x32_f16(false, Au.v, false, B0, (short)0, c0, false, false);
        c1 = __builtin_amdgcn_wmma_f32_16x16x32_f16(false, Au.v, false, B1, (short)0, c1, false, false);
        c2 = __builtin_amdgcn_wmma_f32_16x16x32_f16(false, Au.v, false, B2, (short)0, c2, false, false);
        c3 = __builtin_amdgcn_wmma_f32_16x16x32_f16(false, Au.v, false, B3, (short)0, c3, false, false);
    }

    // ---- spill C tiles (+bias) to LDS. C layout: VGPR r, lane L -> M = r + 8*(L/16), N = L%16
    {
        union { v8f v; float f[8]; } u0, u1, u2, u3;
        u0.v = c0; u1.v = c1; u2.v = c2; u3.v = c3;
#pragma unroll
        for (int r = 0; r < 8; ++r) {
            float* dst = &logits_s[(wave * 16 + hi * 8 + r) * LDS_STR];
            dst[lo]      = u0.f[r] + bb0;
            dst[16 + lo] = u1.f[r] + bb1;
            dst[32 + lo] = u2.f[r] + bb2;
            dst[48 + lo] = u3.f[r] + bb3;
        }
    }
    __syncthreads();

    // ---- per-row softmax + stable top-8 (one thread per row; threads 0..127) ----
    if (tid < ROWS_PB) {
        float* l = &logits_s[tid * LDS_STR];

        float mx = -3.4e38f;
        for (int e = 0; e < NEXP; ++e) mx = fmaxf(mx, l[e]);

        float Z = 0.f;
        for (int e = 0; e < NEXP; ++e) { float p = __expf(l[e] - mx); l[e] = p; Z += p; }

        unsigned long long used = 0ull;
        float sel[TOPK]; int selid[TOPK];
        float ssum = 0.f;
        for (int k = 0; k < TOPK; ++k) {
            float bv = -1.f; int bi = 0;
            for (int e = 0; e < NEXP; ++e) {
                float p = l[e];
                if (!((used >> e) & 1ull) && p > bv) { bv = p; bi = e; }  // ties -> lowest index
            }
            used |= 1ull << bi;
            sel[k] = bv; selid[k] = bi; ssum += bv;
            atomicAdd(&hist_s[bi], 1u);
        }

        // gate_k = (p_k/Z) / (sum_topk/Z + eps) = p_k / (ssum + eps*Z)
        const float inv = 1.0f / (ssum + 1e-8f * Z);
        const size_t r = (size_t)blockIdx.x * ROWS_PB + tid;
        for (int k = 0; k < TOPK; ++k) {
            out_idx [r * TOPK + k] = (float)selid[k];
            out_gate[r * TOPK + k] = sel[k] * inv;
        }
    }
    __syncthreads();

    if (tid < NEXP) atomicAdd(&g_counts[tid], hist_s[tid]);
}

// ---------------- finalize: EMA usage + unbiased variance ----------------

__global__ __launch_bounds__(64) void finalize_k(const unsigned* __restrict__ counts,
                                                 const float* __restrict__ expert_usage,
                                                 float* __restrict__ out_var) {
    __shared__ float u[NEXP];
    const int e = threadIdx.x;
    float usage = (float)counts[e] * (1.0f / (float)N_TOK);
    u[e] = 0.95f * expert_usage[e] + 0.05f * usage;
    __syncthreads();
    if (e == 0) {
        float m = 0.f;
        for (int i = 0; i < NEXP; ++i) m += u[i];
        m *= (1.0f / (float)NEXP);
        float v = 0.f;
        for (int i = 0; i < NEXP; ++i) { float d = u[i] - m; v += d * d; }
        out_var[0] = v * (1.0f / (float)(NEXP - 1));
    }
}

// ---------------- launch ----------------

extern "C" void kernel_launch(void* const* d_in, const int* in_sizes, int n_in,
                              void* d_out, int out_size, void* d_ws, size_t ws_size,
                              hipStream_t stream) {
    const float* x  = (const float*)d_in[0];
    const float* W  = (const float*)d_in[1];
    const float* b  = (const float*)d_in[2];
    const float* eu = (const float*)d_in[3];

    const int N = in_sizes[0] / DIM;   // 262144

    unsigned* counts = (unsigned*)d_ws;
    _Float16* Wh     = (_Float16*)((char*)d_ws + 256);

    float* out       = (float*)d_out;
    float* out_idx   = out;
    float* out_gate  = out + (size_t)N * TOPK;
    float* out_var   = out + (size_t)N * TOPK * 2;

    zero_counts_k<<<1, 64, 0, stream>>>(counts);
    convert_w_k<<<(NEXP * DIM) / 256, 256, 0, stream>>>(W, Wh);
    moe_gate_k<<<N / ROWS_PB, 256, 0, stream>>>(x, Wh, b, out_idx, out_gate, counts);
    finalize_k<<<1, 64, 0, stream>>>(counts, eu, out_var);
}
